// GAT_1288490189412
// MI455X (gfx1250) — matrix operand order repaired and checked by reading, not compiled, not run on hardware
//
#include <hip/hip_runtime.h>
#include <hip/hip_bf16.h>
#include <math.h>

// ---------------------------------------------------------------------------
// GAT (3-layer GATConv, HEADS=4, C=32) for MI455X / gfx1250 (wave32, WMMA).
// GEMM: v_wmma_f32_16x16x32_bf16 with hi/lo bf16 split (3 WMMAs / K32 step,
// ~fp32 accuracy).  A tile converted once per block into LDS (320KB/WGP),
// waves read fragments via ds_load; C staged in LDS for coalesced writeout
// and fused alpha_s/alpha_d attention dots.  Edge phase: L2-resident
// scatter/gather with float atomics (h = 51.2MB fits in 192MB L2).
// ---------------------------------------------------------------------------

typedef __attribute__((ext_vector_type(16))) __bf16 v16bf;
typedef __attribute__((ext_vector_type(8)))  __bf16 v8bf;
typedef __attribute__((ext_vector_type(8)))  float  v8f;
typedef __attribute__((ext_vector_type(4)))  float  v4f;

#define FDIM      128
#define HEADS     4
#define CHN       32
#define NEG_SLOPE 0.2f
#define GAT_EPS   1e-16f

// monotonic float<->uint mapping so atomicMax(uint) == float max
__device__ __forceinline__ unsigned enc_f32(float f) {
    unsigned u = __float_as_uint(f);
    return (u & 0x80000000u) ? ~u : (u | 0x80000000u);
}
__device__ __forceinline__ float dec_f32(unsigned u) {
    unsigned b = (u & 0x80000000u) ? (u & 0x7fffffffu) : ~u;
    return __uint_as_float(b);
}

// ---------------------------------------------------------------------------
// Pack W[l] (128x128 row-major fp32) into per-lane WMMA B-operand layout,
// split into bf16 hi + bf16 lo.  Layout (16-bit B, 32x16; contiguous k order, same as the A fragments):
//   lane<16 : col = 16*t + lane,    K = 32*kb + j      (j=0..15)
//   lane>=16: col = 16*t + lane-16, K = 32*kb + 16 + j
// ---------------------------------------------------------------------------
__global__ void pack_w_kernel(const float* __restrict__ Ws,
                              __bf16* __restrict__ whi,
                              __bf16* __restrict__ wlo) {
    int idx = blockIdx.x * blockDim.x + threadIdx.x;
    if (idx >= 3 * 8 * 4 * 32) return;
    int lane = idx & 31;
    int kb   = (idx >> 5) & 3;
    int t    = (idx >> 7) & 7;
    int l    = idx >> 10;
    int col   = t * 16 + (lane & 15);
    int kbase = kb * 32 + ((lane & 16) ? 16 : 0);
    const float* W = Ws + (size_t)l * FDIM * FDIM;
    __bf16* dhi = whi + (size_t)idx * 16;
    __bf16* dlo = wlo + (size_t)idx * 16;
#pragma unroll
    for (int j = 0; j < 16; ++j) {
        float v   = W[(size_t)(kbase + j) * FDIM + col];
        __bf16 hi = (__bf16)v;
        dhi[j] = hi;
        dlo[j] = (__bf16)(v - (float)hi);
    }
}

// ---------------------------------------------------------------------------
// Fused GEMM + attention dots.  Block = 256 threads = 8 waves; block computes
// a 16x128 tile of h = in @ W; wave w owns cols 16w..16w+15.
//   phase 1: all 256 threads load the 16x128 fp32 A tile (coalesced b128),
//            split to bf16 hi/lo, store to LDS (once per block, not per wave)
//   phase 2: each wave ds_loads its A fragments (contiguous k order, as in B) and
//            runs 4 K-blocks x 3 WMMAs
//   phase 3: C tile staged in LDS
//   phase 4: coalesced h writeout + fused alpha_s/alpha_d dot products
// ---------------------------------------------------------------------------
__global__ void __launch_bounds__(256)
gemm_wmma_kernel(const float* __restrict__ in,
                 const __bf16* __restrict__ whi,
                 const __bf16* __restrict__ wlo,
                 const float* __restrict__ asrc,
                 const float* __restrict__ adst,
                 float* __restrict__ h,
                 float* __restrict__ als,
                 float* __restrict__ ald) {
    __shared__ __align__(16) __bf16 sAhi[16 * FDIM];
    __shared__ __align__(16) __bf16 sAlo[16 * FDIM];
    __shared__ __align__(16) float  sC[16 * FDIM];

    const int t       = threadIdx.x;
    const int rowbase = blockIdx.x * 16;

    // ---- phase 1: cooperative A-tile load + hi/lo split ----
    {
        int idx = t * 8;                  // element index within 16x128 tile
        int r   = idx >> 7;
        int c0  = idx & 127;
        const float* src = in + (size_t)(rowbase + r) * FDIM + c0;
        v4f f0 = *(const v4f*)(src);
        v4f f1 = *(const v4f*)(src + 4);
#pragma unroll
        for (int j = 0; j < 8; ++j) {
            float v   = (j < 4) ? f0[j] : f1[j - 4];
            __bf16 hi = (__bf16)v;
            sAhi[idx + j] = hi;
            sAlo[idx + j] = (__bf16)(v - (float)hi);
        }
    }
    __syncthreads();

    const int wave = t >> 5;
    const int lane = t & 31;
    const int arow = lane & 15;

    // ---- phase 2: WMMA over K=128 ----
    v8f c = {};
#pragma unroll
    for (int kb = 0; kb < 4; ++kb) {
        int kofs = kb * 32 + ((lane & 16) ? 16 : 0);
        // A fragment: j=0..15 -> K=kofs..kofs+15
        union { v16bf v; v8bf p[2]; } ahi, alo;
        ahi.p[0] = *(const v8bf*)(sAhi + arow * FDIM + kofs);
        ahi.p[1] = *(const v8bf*)(sAhi + arow * FDIM + kofs + 8);
        alo.p[0] = *(const v8bf*)(sAlo + arow * FDIM + kofs);
        alo.p[1] = *(const v8bf*)(sAlo + arow * FDIM + kofs + 8);
        size_t bofs = ((size_t)(wave * 4 + kb) * 32 + lane) * 16;
        v16bf bhi = *(const v16bf*)(whi + bofs);
        v16bf blo = *(const v16bf*)(wlo + bofs);
        // D = Ahi*Bhi + Ahi*Blo + Alo*Bhi + C   (~fp32 via bf16 split)
        c = __builtin_amdgcn_wmma_f32_16x16x32_bf16(false, ahi.v, false, bhi,
                                                    (short)0, c, false, false);
        c = __builtin_amdgcn_wmma_f32_16x16x32_bf16(false, ahi.v, false, blo,
                                                    (short)0, c, false, false);
        c = __builtin_amdgcn_wmma_f32_16x16x32_bf16(false, alo.v, false, bhi,
                                                    (short)0, c, false, false);
    }

    // ---- phase 3: stage C tile in LDS ----
    {
        // C/D layout: vgpr i -> M = i (+8 upper half-wave), N = lane&15
        int col = wave * 16 + (lane & 15);
        int rb  = (lane & 16) ? 8 : 0;
#pragma unroll
        for (int i = 0; i < 8; ++i)
            sC[(rb + i) * FDIM + col] = c[i];
    }
    __syncthreads();

    // ---- phase 4a: coalesced h writeout ----
    {
        int idx = t * 8;
        int r   = idx >> 7;
        int c0  = idx & 127;
        float* dst = h + (size_t)(rowbase + r) * FDIM + c0;
        v4f o0, o1;
#pragma unroll
        for (int j = 0; j < 4; ++j) { o0[j] = sC[idx + j]; o1[j] = sC[idx + 4 + j]; }
        *(v4f*)(dst)     = o0;
        *(v4f*)(dst + 4) = o1;
    }
    // ---- phase 4b: fused attention dots (threads 0..63: 16 rows x 4 heads) --
    if (t < 64) {
        int r  = t >> 2;
        int hh = t & 3;
        const float* hp  = sC + r * FDIM + hh * CHN;
        const float* wsv = asrc + hh * CHN;
        const float* wdv = adst + hh * CHN;
        float ss = 0.f, sd = 0.f;
#pragma unroll
        for (int c2 = 0; c2 < CHN; ++c2) {
            float v = hp[c2];
            ss += v * wsv[c2];
            sd += v * wdv[c2];
        }
        size_t o = (size_t)(rowbase + r) * HEADS + hh;
        als[o] = ss;
        ald[o] = sd;
    }
}

__device__ __forceinline__ void edge_sd(const long long* ei, int e, int E,
                                        int& s, int& d) {
    if (e < E) { s = (int)ei[e]; d = (int)ei[(size_t)E + e]; }
    else       { s = d = e - E; }
}

// pass 1: m[dst,h] = max over incoming edges of leaky_relu(as[src]+ad[dst])
__global__ void edge_max_kernel(const long long* __restrict__ ei,
                                const float* __restrict__ als,
                                const float* __restrict__ ald,
                                unsigned* __restrict__ menc, int E, int nN) {
    int e = blockIdx.x * blockDim.x + threadIdx.x;
    if (e >= E + nN) return;
    int s, d; edge_sd(ei, e, E, s, d);
    v4f as = *(const v4f*)(als + (size_t)s * 4);
    v4f ad = *(const v4f*)(ald + (size_t)d * 4);
#pragma unroll
    for (int hh = 0; hh < 4; ++hh) {
        float ev = as[hh] + ad[hh];
        ev = ev > 0.f ? ev : NEG_SLOPE * ev;
        atomicMax(menc + (size_t)d * 4 + hh, enc_f32(ev));
    }
}

// pass 2: denom[dst,h] = sum exp(e - m[dst,h])
__global__ void edge_sum_kernel(const long long* __restrict__ ei,
                                const float* __restrict__ als,
                                const float* __restrict__ ald,
                                const unsigned* __restrict__ menc,
                                float* __restrict__ denom, int E, int nN) {
    int e = blockIdx.x * blockDim.x + threadIdx.x;
    if (e >= E + nN) return;
    int s, d; edge_sd(ei, e, E, s, d);
    v4f as = *(const v4f*)(als + (size_t)s * 4);
    v4f ad = *(const v4f*)(ald + (size_t)d * 4);
#pragma unroll
    for (int hh = 0; hh < 4; ++hh) {
        float ev = as[hh] + ad[hh];
        ev = ev > 0.f ? ev : NEG_SLOPE * ev;
        float m = dec_f32(menc[(size_t)d * 4 + hh]);
        atomicAdd(denom + (size_t)d * 4 + hh, __expf(ev - m));
    }
}

// pass 3: acc[dst] += alpha * h[src]   (one wave32 per edge: lane = 4 channels)
__global__ void __launch_bounds__(256)
edge_agg_kernel(const long long* __restrict__ ei,
                const float* __restrict__ als,
                const float* __restrict__ ald,
                const unsigned* __restrict__ menc,
                const float* __restrict__ denom,
                const float* __restrict__ h,
                float* __restrict__ acc, int E, int nN) {
    int e = blockIdx.x * 8 + (threadIdx.x >> 5);
    if (e >= E + nN) return;
    int lane = threadIdx.x & 31;
    int s, d; edge_sd(ei, e, E, s, d);
    int hh = lane >> 3;                      // head of this lane's 4 channels
    float ev = als[(size_t)s * 4 + hh] + ald[(size_t)d * 4 + hh];
    ev = ev > 0.f ? ev : NEG_SLOPE * ev;
    float m  = dec_f32(menc[(size_t)d * 4 + hh]);
    float w  = __expf(ev - m) / (denom[(size_t)d * 4 + hh] + GAT_EPS);
    v4f hv = *(const v4f*)(h + (size_t)s * FDIM + lane * 4);
    float* ap = acc + (size_t)d * FDIM + lane * 4;
    atomicAdd(ap + 0, w * hv[0]);
    atomicAdd(ap + 1, w * hv[1]);
    atomicAdd(ap + 2, w * hv[2]);
    atomicAdd(ap + 3, w * hv[3]);
}

__global__ void bias_kernel(const float* __restrict__ acc,
                            const float* __restrict__ b,
                            float* __restrict__ out, int do_relu, int total) {
    int i = blockIdx.x * blockDim.x + threadIdx.x;
    if (i >= total) return;
    float v = acc[i] + b[i & (FDIM - 1)];
    if (do_relu) v = v > 0.f ? v : 0.f;
    out[i] = v;
}

// ---------------------------------------------------------------------------
extern "C" void kernel_launch(void* const* d_in, const int* in_sizes, int n_in,
                              void* d_out, int out_size, void* d_ws,
                              size_t ws_size, hipStream_t stream) {
    const float*     x       = (const float*)d_in[0];      // [N,128] f32
    const long long* ei      = (const long long*)d_in[1];  // [2,E] int64
    const float*     Ws      = (const float*)d_in[2];      // [3,128,128]
    const float*     att_src = (const float*)d_in[3];      // [3,4,32]
    const float*     att_dst = (const float*)d_in[4];      // [3,4,32]
    const float*     bias    = (const float*)d_in[5];      // [3,128]
    float*           out     = (float*)d_out;

    const int N  = in_sizes[0] / FDIM;   // 100000 (multiple of 16)
    const int E  = in_sizes[1] / 2;      // 1600000
    const int EN = E + N;                // edges + self loops

    // ---- workspace layout (~109 MB) ----
    char* p = (char*)d_ws;
    float* bufA = (float*)p; p += sizeof(float) * (size_t)N * FDIM;   // h
    float* bufB = (float*)p; p += sizeof(float) * (size_t)N * FDIM;   // acc / x'
    float*    als   = (float*)p;    p += sizeof(float) * (size_t)N * HEADS;
    float*    ald   = (float*)p;    p += sizeof(float) * (size_t)N * HEADS;
    unsigned* menc  = (unsigned*)p; p += sizeof(unsigned) * (size_t)N * HEADS;
    float*    denom = (float*)p;    p += sizeof(float) * (size_t)N * HEADS;
    __bf16* whi = (__bf16*)p; p += sizeof(__bf16) * 3 * 16384;
    __bf16* wlo = (__bf16*)p; p += sizeof(__bf16) * 3 * 16384;

    // pack all 3 weight matrices into WMMA B layout (hi/lo bf16)
    pack_w_kernel<<<(3 * 8 * 4 * 32 + 255) / 256, 256, 0, stream>>>(Ws, whi, wlo);

    const float* in = x;
    for (int l = 0; l < 3; ++l) {
        // fused GEMM (WMMA bf16 split) + attention dots
        gemm_wmma_kernel<<<N / 16, 256, 0, stream>>>(
            in, whi + (size_t)l * 16384, wlo + (size_t)l * 16384,
            att_src + l * FDIM, att_dst + l * FDIM, bufA, als, ald);
        // init segment buffers (input buffer is dead after the GEMM -> reuse
        // it as this layer's accumulator; stream order makes this safe)
        hipMemsetAsync(menc, 0, sizeof(unsigned) * (size_t)N * HEADS, stream);
        hipMemsetAsync(denom, 0, sizeof(float) * (size_t)N * HEADS, stream);
        hipMemsetAsync(bufB, 0, sizeof(float) * (size_t)N * FDIM, stream);
        // segment softmax + aggregation (L2-resident scatter/gather)
        edge_max_kernel<<<(EN + 255) / 256, 256, 0, stream>>>(ei, als, ald,
                                                              menc, E, N);
        edge_sum_kernel<<<(EN + 255) / 256, 256, 0, stream>>>(ei, als, ald,
                                                              menc, denom, E, N);
        edge_agg_kernel<<<(EN + 7) / 8, 256, 0, stream>>>(ei, als, ald, menc,
                                                          denom, bufA, bufB, E, N);
        // bias (+ReLU between layers); last layer writes d_out
        float* dst = (l == 2) ? out : bufB;
        bias_kernel<<<((size_t)N * FDIM + 255) / 256, 256, 0, stream>>>(
            bufB, bias + l * FDIM, dst, (l < 2) ? 1 : 0, N * FDIM);
        in = bufB;
    }
}
